// Transformer_2233382994660
// MI455X (gfx1250) — matrix operand order repaired and checked by reading, not compiled, not run on hardware
//
#include <hip/hip_runtime.h>

// ---------------------------------------------------------------------------
// DeepViT block (4 layers) for MI455X / gfx1250, wave32 + WMMA.
// All GEMMs run on v_wmma_f32_16x16x32_bf16 using split-bf16 (bf16x3)
// operands for ~fp32 accuracy at ~8/3 the f32-WMMA instruction throughput.
// ---------------------------------------------------------------------------

typedef __attribute__((ext_vector_type(16))) __bf16          v16bf;
typedef __attribute__((ext_vector_type(8)))  float           v8f;
typedef __attribute__((ext_vector_type(8)))  unsigned short  v8us;

union FragU {
    v16bf v;
    v8us  h[2];
};

__device__ __forceinline__ unsigned short f2bf(float f) {
    union { float f; unsigned u; } v; v.f = f;
    unsigned r = v.u + 0x7FFFu + ((v.u >> 16) & 1u);   // round-to-nearest-even
    return (unsigned short)(r >> 16);
}
__device__ __forceinline__ float bf2f(unsigned short h) {
    union { unsigned u; float f; } v; v.u = ((unsigned)h) << 16;
    return v.f;
}

// ---------------------------------------------------------------------------
// Tiled GEMM: C[M,N] = act( alpha * A[M,K] @ B + bias (+ old C if ADD) )
//   BT=false: B is [K,N] row-major (ldb = row stride of B)
//   BT=true : B is [N,K] row-major (C = A @ B^T), contiguous K per output col
// Block: 128x128 tile, 256 threads = 8 waves in 2(M) x 4(N); each wave does
// 4x2 WMMA 16x16 tiles, 3 wmma per tile per K=32 step (hi*hi + hi*lo + lo*hi).
// z-dim of the grid indexes independent (batch,head) problems via strides.
// Requires: M % 128 == 0, K % 32 == 0. N may be ragged (masked).
// ACT: 0 = none, 1 = exact GELU, 2 = tanh.  ADD: old C added before ACT (residual).
// ---------------------------------------------------------------------------
template <bool BT, int ACT, bool ADD>
__global__ __launch_bounds__(256)
void gemm_bf16x3(const float* __restrict__ A, long lda, long zsa,
                 const float* __restrict__ B, long ldb, long zsb,
                 float* __restrict__ C, long ldc, long zsc,
                 const float* __restrict__ bias,
                 int M, int N, int K, float alpha)
{
    __shared__ __align__(16) unsigned short sA[2][128][40]; // [hi/lo][m][k] pad->80B rows
    __shared__ __align__(16) unsigned short sB[2][128][40]; // [hi/lo][n][k]

    const int tid  = threadIdx.x;
    const int lane = tid & 31;
    const int wave = tid >> 5;
    const int wm   = wave >> 2;   // 0..1  -> 64 rows
    const int wn   = wave & 3;    // 0..3  -> 32 cols
    const int bm   = blockIdx.y * 128;
    const int bn   = blockIdx.x * 128;
    const int z    = blockIdx.z;

    A += (long)z * zsa;
    B += (long)z * zsb;
    C += (long)z * zsc;

    v8f acc[4][2];
#pragma unroll
    for (int i = 0; i < 4; ++i)
#pragma unroll
        for (int j = 0; j < 2; ++j)
#pragma unroll
            for (int e = 0; e < 8; ++e) acc[i][j][e] = 0.0f;

    // Staging: each thread converts 16 contiguous f32 -> hi/lo bf16 of one row
    const int lrow = tid >> 1;          // 0..127
    const int lk   = (tid & 1) * 16;    // 0 or 16

    for (int k0 = 0; k0 < K; k0 += 32) {
        { // A tile (rows always in range: M % 128 == 0)
            const float* ap = A + (long)(bm + lrow) * lda + (k0 + lk);
#pragma unroll
            for (int u = 0; u < 16; ++u) {
                float x = ap[u];
                unsigned short hi = f2bf(x);
                sA[0][lrow][lk + u] = hi;
                sA[1][lrow][lk + u] = f2bf(x - bf2f(hi));
            }
        }
        { // B tile, stored K-contiguous per output column n
            const int n = bn + lrow;
#pragma unroll
            for (int u = 0; u < 16; ++u) {
                const int kk = k0 + lk + u;
                float x = 0.0f;
                if (n < N) x = BT ? B[(long)n * ldb + kk] : B[(long)kk * ldb + n];
                unsigned short hi = f2bf(x);
                sB[0][lrow][lk + u] = hi;
                sB[1][lrow][lk + u] = f2bf(x - bf2f(hi));
            }
        }
        __syncthreads();

        // B fragments: lane<16 -> col lane, K 0..7 & 16..23 ; lane>=16 -> +8
        FragU b_hi[2], b_lo[2];
        {
            const int bn_l = lane & 15;
            const int bc   = (lane >> 4) * 8;
#pragma unroll
            for (int j = 0; j < 2; ++j) {
                const unsigned short* r0 = &sB[0][wn * 32 + j * 16 + bn_l][0];
                const unsigned short* r1 = &sB[1][wn * 32 + j * 16 + bn_l][0];
                b_hi[j].h[0] = *(const v8us*)(r0 + bc);
                b_hi[j].h[1] = *(const v8us*)(r0 + bc + 16);
                b_lo[j].h[0] = *(const v8us*)(r1 + bc);
                b_lo[j].h[1] = *(const v8us*)(r1 + bc + 16);
            }
        }
        // A fragments: lane<16 -> row lane, K 0..7 & 16..23 ; lane>=16 -> +8
        const int am = lane & 15;
        const int ac = (lane >> 4) * 8;
#pragma unroll
        for (int i = 0; i < 4; ++i) {
            FragU a_hi, a_lo;
            const unsigned short* r0 = &sA[0][wm * 64 + i * 16 + am][0];
            const unsigned short* r1 = &sA[1][wm * 64 + i * 16 + am][0];
            a_hi.h[0] = *(const v8us*)(r0 + ac);
            a_hi.h[1] = *(const v8us*)(r0 + ac + 16);
            a_lo.h[0] = *(const v8us*)(r1 + ac);
            a_lo.h[1] = *(const v8us*)(r1 + ac + 16);
#pragma unroll
            for (int j = 0; j < 2; ++j) {
                acc[i][j] = __builtin_amdgcn_wmma_f32_16x16x32_bf16(
                    false, a_hi.v, false, b_hi[j].v, (short)0, acc[i][j], false, false);
                acc[i][j] = __builtin_amdgcn_wmma_f32_16x16x32_bf16(
                    false, a_hi.v, false, b_lo[j].v, (short)0, acc[i][j], false, false);
                acc[i][j] = __builtin_amdgcn_wmma_f32_16x16x32_bf16(
                    false, a_lo.v, false, b_hi[j].v, (short)0, acc[i][j], false, false);
            }
        }
        __syncthreads();
    }

    // Epilogue: D layout — VGPR e: row = e + 8*(lane>=16), col = lane&15
    const int col0 = lane & 15;
    const int rowh = (lane >> 4) * 8;
#pragma unroll
    for (int j = 0; j < 2; ++j) {
        const int gcol = bn + wn * 32 + j * 16 + col0;
        if (gcol >= N) continue;
        const float bv = bias ? bias[gcol] : 0.0f;
#pragma unroll
        for (int i = 0; i < 4; ++i) {
#pragma unroll
            for (int e = 0; e < 8; ++e) {
                const int  grow = bm + wm * 64 + i * 16 + rowh + e;
                const long off  = (long)grow * ldc + gcol;
                float v = alpha * acc[i][j][e] + bv;
                if (ADD) v += C[off];
                if (ACT == 1) v = 0.5f * v * (1.0f + erff(v * 0.70710678118654752f));
                else if (ACT == 2) v = tanhf(v);
                C[off] = v;
            }
        }
    }
}

// ---------------------------------------------------------------------------
// Row LayerNorm: one 256-thread block per row of `cols` (cols = 512 here)
// ---------------------------------------------------------------------------
__global__ __launch_bounds__(256)
void ln_rows(const float* __restrict__ x, const float* __restrict__ g,
             const float* __restrict__ b, float* __restrict__ y, int cols)
{
    const long row = blockIdx.x;
    const float* xr = x + row * cols;
    float* yr = y + row * cols;
    __shared__ float red[256];

    float s = 0.0f;
    for (int c = threadIdx.x; c < cols; c += 256) s += xr[c];
    red[threadIdx.x] = s;
    __syncthreads();
    for (int st = 128; st > 0; st >>= 1) {
        if (threadIdx.x < st) red[threadIdx.x] += red[threadIdx.x + st];
        __syncthreads();
    }
    const float mu = red[0] / (float)cols;
    __syncthreads();

    float s2 = 0.0f;
    for (int c = threadIdx.x; c < cols; c += 256) {
        const float d = xr[c] - mu;
        s2 += d * d;
    }
    red[threadIdx.x] = s2;
    __syncthreads();
    for (int st = 128; st > 0; st >>= 1) {
        if (threadIdx.x < st) red[threadIdx.x] += red[threadIdx.x + st];
        __syncthreads();
    }
    const float rstd = rsqrtf(red[0] / (float)cols + 1e-5f);

    for (int c = threadIdx.x; c < cols; c += 256)
        yr[c] = (xr[c] - mu) * rstd * g[c] + b[c];
}

// ---------------------------------------------------------------------------
// Row softmax over 1024 columns, in place. One block per row.
// ---------------------------------------------------------------------------
__global__ __launch_bounds__(256)
void softmax_rows(float* __restrict__ s, int cols)
{
    float* r = s + (long)blockIdx.x * cols;
    __shared__ float red[256];

    float m = -3.402823466e+38f;
    for (int c = threadIdx.x; c < cols; c += 256) m = fmaxf(m, r[c]);
    red[threadIdx.x] = m;
    __syncthreads();
    for (int st = 128; st > 0; st >>= 1) {
        if (threadIdx.x < st) red[threadIdx.x] = fmaxf(red[threadIdx.x], red[threadIdx.x + st]);
        __syncthreads();
    }
    m = red[0];
    __syncthreads();

    float sum = 0.0f;
    for (int c = threadIdx.x; c < cols; c += 256) {
        const float e = expf(r[c] - m);
        r[c] = e;
        sum += e;
    }
    red[threadIdx.x] = sum;
    __syncthreads();
    for (int st = 128; st > 0; st >>= 1) {
        if (threadIdx.x < st) red[threadIdx.x] += red[threadIdx.x + st];
        __syncthreads();
    }
    const float inv = 1.0f / red[0];
    for (int c = threadIdx.x; c < cols; c += 256) r[c] *= inv;
}

// ---------------------------------------------------------------------------
// Re-attention: attn'[g,i,j] = sum_h attn[h,i,j] * rw[h,g], then LayerNorm
// over the head axis (8 values) with gain/bias. In place: each thread owns
// one (i,j) column across all 8 heads (head stride = 1024*1024).
// ---------------------------------------------------------------------------
__global__ __launch_bounds__(256)
void reattn_mix_ln(float* __restrict__ s, const float* __restrict__ rw,
                   const float* __restrict__ rg, const float* __restrict__ rb)
{
    __shared__ float w[64], g[8], bb[8];
    if (threadIdx.x < 64) w[threadIdx.x] = rw[threadIdx.x];
    if (threadIdx.x < 8) { g[threadIdx.x] = rg[threadIdx.x]; bb[threadIdx.x] = rb[threadIdx.x]; }
    __syncthreads();

    const long idx = (long)blockIdx.x * 256 + threadIdx.x;   // 0 .. 1024*1024-1
    const long HS  = 1024l * 1024l;

    float a[8];
#pragma unroll
    for (int h = 0; h < 8; ++h) a[h] = s[(long)h * HS + idx];

    float m[8];
    float mu = 0.0f;
#pragma unroll
    for (int gg = 0; gg < 8; ++gg) {
        float t = 0.0f;
#pragma unroll
        for (int h = 0; h < 8; ++h) t += a[h] * w[h * 8 + gg];   // 'bhij,hg->bgij'
        m[gg] = t;
        mu += t;
    }
    mu *= 0.125f;
    float var = 0.0f;
#pragma unroll
    for (int gg = 0; gg < 8; ++gg) { const float d = m[gg] - mu; var += d * d; }
    var *= 0.125f;
    const float rstd = rsqrtf(var + 1e-5f);
#pragma unroll
    for (int gg = 0; gg < 8; ++gg)
        s[(long)gg * HS + idx] = (m[gg] - mu) * rstd * g[gg] + bb[gg];
}

// ---------------------------------------------------------------------------
// Host-side orchestration. Workspace (floats):
//   hn       4096*512    =  8 MiB
//   qkv      4096*1536   = 24 MiB
//   scores   8*1024*1024 = 32 MiB   (per-batch, reused b=0..3)
//   attn_out 4096*512    =  8 MiB
//   hid      4096*2048   = 32 MiB
// total ~104 MiB of d_ws. d_out doubles as the residual-stream x buffer.
// ---------------------------------------------------------------------------
extern "C" void kernel_launch(void* const* d_in, const int* in_sizes, int n_in,
                              void* d_out, int out_size, void* d_ws, size_t ws_size,
                              hipStream_t stream)
{
    (void)in_sizes; (void)n_in; (void)out_size; (void)ws_size;

    const float* x_in   = (const float*)d_in[0];
    const float* ln1_g  = (const float*)d_in[1];
    const float* ln1_b  = (const float*)d_in[2];
    const float* w_qkv  = (const float*)d_in[3];
    const float* rw     = (const float*)d_in[4];
    const float* rln_g  = (const float*)d_in[5];
    const float* rln_b  = (const float*)d_in[6];
    const float* w_out  = (const float*)d_in[7];
    const float* b_out  = (const float*)d_in[8];
    const float* ln2_g  = (const float*)d_in[9];
    const float* ln2_b  = (const float*)d_in[10];
    const float* w1     = (const float*)d_in[11];
    const float* b1     = (const float*)d_in[12];
    const float* w2     = (const float*)d_in[13];
    const float* b2     = (const float*)d_in[14];

    float* x = (float*)d_out;                         // residual stream, updated in place

    float* hn       = (float*)d_ws;
    float* qkv      = hn       + 4096l * 512;
    float* scores   = qkv      + 4096l * 1536;
    float* attn_out = scores   + 8l * 1024 * 1024;
    float* hid      = attn_out + 4096l * 512;

    hipMemcpyAsync(x, x_in, 4096l * 512 * sizeof(float), hipMemcpyDeviceToDevice, stream);

    const float SCALE = 0.125f;   // 64^-0.5

    for (int l = 0; l < 4; ++l) {
        // ---- attention branch -------------------------------------------
        ln_rows<<<4096, 256, 0, stream>>>(x, ln1_g + l * 512, ln1_b + l * 512, hn, 512);

        {   // qkv = hn @ w_qkv[l]           (4096 x 1536 x 512)
            dim3 grd(1536 / 128, 4096 / 128, 1);
            gemm_bf16x3<false, 0, false><<<grd, 256, 0, stream>>>(
                hn, 512, 0, w_qkv + (long)l * 512 * 1536, 1536, 0,
                qkv, 1536, 0, nullptr, 4096, 1536, 512, 1.0f);
        }

        for (int b = 0; b < 4; ++b) {
            const float* qb = qkv + (long)b * 1024 * 1536;
            {   // scores[h] = SCALE * Q_h @ K_h^T   (NT GEMM, z = head)
                dim3 grd(1024 / 128, 1024 / 128, 8);
                gemm_bf16x3<true, 0, false><<<grd, 256, 0, stream>>>(
                    qb, 1536, 64, qb + 512, 1536, 64,
                    scores, 1024, 1024l * 1024, nullptr, 1024, 1024, 64, SCALE);
            }
            softmax_rows<<<8 * 1024, 256, 0, stream>>>(scores, 1024);
            reattn_mix_ln<<<4096, 256, 0, stream>>>(scores, rw + l * 64,
                                                    rln_g + l * 8, rln_b + l * 8);
            {   // attn_out[b,:,h*64:(h+1)*64] = attn'_h @ V_h   (z = head)
                dim3 grd(1, 1024 / 128, 8);
                gemm_bf16x3<false, 0, false><<<grd, 256, 0, stream>>>(
                    scores, 1024, 1024l * 1024, qb + 1024, 1536, 64,
                    attn_out + (long)b * 1024 * 512, 512, 64,
                    nullptr, 1024, 64, 1024, 1.0f);
            }
        }

        {   // x += attn_out @ w_out[l] + b_out[l]
            dim3 grd(512 / 128, 4096 / 128, 1);
            gemm_bf16x3<false, 0, true><<<grd, 256, 0, stream>>>(
                attn_out, 512, 0, w_out + (long)l * 512 * 512, 512, 0,
                x, 512, 0, b_out + l * 512, 4096, 512, 512, 1.0f);
        }

        // ---- FFN branch --------------------------------------------------
        ln_rows<<<4096, 256, 0, stream>>>(x, ln2_g + l * 512, ln2_b + l * 512, hn, 512);

        {   // hid = gelu(hn @ w1[l] + b1[l])
            dim3 grd(2048 / 128, 4096 / 128, 1);
            gemm_bf16x3<false, 1, false><<<grd, 256, 0, stream>>>(
                hn, 512, 0, w1 + (long)l * 512 * 2048, 2048, 0,
                hid, 2048, 0, b1 + l * 2048, 4096, 2048, 512, 1.0f);
        }
        {   // x = tanh(x + hid @ w2[l] + b2[l])
            dim3 grd(512 / 128, 4096 / 128, 1);
            gemm_bf16x3<false, 2, true><<<grd, 256, 0, stream>>>(
                hid, 2048, 0, w2 + (long)l * 2048 * 512, 512, 0,
                x, 512, 0, b2 + l * 512, 4096, 512, 2048, 1.0f);
        }
    }
}
